// GraphLearned_87170656239627
// MI455X (gfx1250) — matrix operand order repaired and checked by reading, not compiled, not run on hardware
//
#include <hip/hip_runtime.h>
#include <hip/hip_bf16.h>
#include <math.h>

#define NN 4096
#define DD 128

typedef __attribute__((ext_vector_type(16))) __bf16 v16bf;
typedef __attribute__((ext_vector_type(8)))  __bf16 v8bf;
typedef __attribute__((ext_vector_type(8)))  float  v8f;

__device__ __forceinline__ v8f wmma_bf16(v16bf a, v16bf b, v8f c) {
  return __builtin_amdgcn_wmma_f32_16x16x32_bf16(false, a, false, b, (short)0, c, false, false);
}

// ---------------- selection kernels ----------------
// 16th smallest of each row of dis (exact via bitwise binary search; dis > 0)
__global__ void gl_kth_row(const float* __restrict__ dis, float* __restrict__ rowk) {
  __shared__ unsigned int sdat[NN];
  __shared__ int scnt;
  int row = blockIdx.x;
  const float* r = dis + (size_t)row * NN;
  for (int i = threadIdx.x; i < NN; i += blockDim.x) sdat[i] = __float_as_uint(r[i]);
  __syncthreads();
  unsigned int lo = 0u, hi = 0x7F800000u;
  for (int it = 0; it < 32 && lo < hi; ++it) {
    unsigned int mid = (lo + hi) >> 1;
    if (threadIdx.x == 0) scnt = 0;
    __syncthreads();
    int c = 0;
    for (int i = threadIdx.x; i < NN; i += blockDim.x) c += (sdat[i] <= mid);
    atomicAdd(&scnt, c);
    __syncthreads();
    int cnt = scnt;
    __syncthreads();
    if (cnt >= 16) hi = mid; else lo = mid + 1u;
  }
  if (threadIdx.x == 0) rowk[row] = __uint_as_float(hi);
}

// median of the 4096 per-row values = mean of order stats 2048 and 2049
__global__ void gl_median(const float* __restrict__ rowk, float* __restrict__ dcut) {
  __shared__ unsigned int sdat[NN];
  __shared__ int scnt;
  for (int i = threadIdx.x; i < NN; i += blockDim.x) sdat[i] = __float_as_uint(rowk[i]);
  __syncthreads();
  float vals[2];
  for (int t = 0; t < 2; ++t) {
    int k = 2048 + t;
    unsigned int lo = 0u, hi = 0x7F800000u;
    for (int it = 0; it < 32 && lo < hi; ++it) {
      unsigned int mid = (lo + hi) >> 1;
      if (threadIdx.x == 0) scnt = 0;
      __syncthreads();
      int c = 0;
      for (int i = threadIdx.x; i < NN; i += blockDim.x) c += (sdat[i] <= mid);
      atomicAdd(&scnt, c);
      __syncthreads();
      int cnt = scnt;
      __syncthreads();
      if (cnt >= k) hi = mid; else lo = mid + 1u;
    }
    vals[t] = __uint_as_float(hi);
  }
  if (threadIdx.x == 0) *dcut = 0.5f * (vals[0] + vals[1]);
}

// deg[i] = 1 + column-sum_j adj[j][i]  (A = adj^T, A_hat = A + I); dinv = deg^{-1/2}
__global__ void gl_coldeg(const float* __restrict__ adj, float* __restrict__ dinv) {
  int i = blockIdx.x * blockDim.x + threadIdx.x;
  float s = 1.0f;
  for (int j = 0; j < NN; ++j) s += adj[(size_t)j * NN + i];
  dinv[i] = rsqrtf(s);
}

// ---------------- WMMA GEMM kernels ----------------
// C[4096,128] = A[4096,128] @ W[128,128].
// W staged whole in LDS as bf16, TRANSPOSED (sWt[n][k]) so a lane's 16 K-values
// are one contiguous 32B v16bf read.
__global__ void gl_gemm_xw(const float* __restrict__ A, const float* __restrict__ W,
                           float* __restrict__ C) {
  __shared__ __bf16 sWt[DD * DD];
  int tid = threadIdx.x;
  for (int t = tid; t < DD * DD / 4; t += 256) {
    float4 v = ((const float4*)W)[t];
    int k = t >> 5;            // flat = t*4; k = flat>>7
    int n = (t & 31) * 4;      // n = flat & 127
    sWt[(n + 0) * DD + k] = (__bf16)v.x;
    sWt[(n + 1) * DD + k] = (__bf16)v.y;
    sWt[(n + 2) * DD + k] = (__bf16)v.z;
    sWt[(n + 3) * DD + k] = (__bf16)v.w;
  }
  __syncthreads();
  int widx = tid >> 5, lane = tid & 31;
  int half = lane >> 4, l = lane & 15;
  int w = blockIdx.x * 8 + widx;
  int i0 = (w >> 3) * 16, j0 = (w & 7) * 16;
  v8f acc = {};
#pragma unroll
  for (int k0 = 0; k0 < DD; k0 += 32) {
    const float* pa = A + (size_t)(i0 + l) * DD + k0;
    float4 a0 = *(const float4*)(pa + 16 * half);
    float4 a1 = *(const float4*)(pa + 16 * half + 4);
    float4 a2 = *(const float4*)(pa + 16 * half + 8);
    float4 a3 = *(const float4*)(pa + 16 * half + 12);
    v16bf a;
    a[0]  = (__bf16)a0.x; a[1]  = (__bf16)a0.y; a[2]  = (__bf16)a0.z; a[3]  = (__bf16)a0.w;
    a[4]  = (__bf16)a1.x; a[5]  = (__bf16)a1.y; a[6]  = (__bf16)a1.z; a[7]  = (__bf16)a1.w;
    a[8]  = (__bf16)a2.x; a[9]  = (__bf16)a2.y; a[10] = (__bf16)a2.z; a[11] = (__bf16)a2.w;
    a[12] = (__bf16)a3.x; a[13] = (__bf16)a3.y; a[14] = (__bf16)a3.z; a[15] = (__bf16)a3.w;
    v16bf b = *(const v16bf*)(sWt + (j0 + l) * DD + k0 + 16 * half);
    acc = wmma_bf16(a, b, acc);
  }
#pragma unroll
  for (int r = 0; r < 8; ++r)
    C[(size_t)(i0 + r + 8 * half) * DD + (j0 + l)] = acc[r];
}

// H = act( dinv[i] * ( (adj^T @ (dinv ⊙ X))[i] + dinv[i]*X[i] ) + b ).
// Double-buffered software pipeline: A-tile via async global->LDS (ASYNCcnt),
// B-slab via coalesced float4 loads -> bf16 TRANSPOSED in LDS (sBt[n][k]).
__global__ void gl_agg(const float* __restrict__ adj, const float* __restrict__ X,
                       const float* __restrict__ dinv, const float* __restrict__ bias,
                       float* __restrict__ H, int do_relu) {
  __shared__ float smem[1024 + 4096];         // 2 x (16x32 f32 A, as sA[m][k]) + 2 x (4096 bf16 B)
  __bf16* sBt = (__bf16*)(smem + 1024);       // sBt[buf][n*32+k]
  int tid  = threadIdx.x;
  int widx = tid >> 5, lane = tid & 31;
  int half = lane >> 4, l = lane & 15;
  int i0 = blockIdx.x * 16, j0 = widx * 16;   // all 8 waves share i0; tj = widx
  int ar = tid >> 4, ac = tid & 15;           // A staging: ar = k row (0..15[+16]), ac = m col

  auto issueA = [&](int k0, int buf) {
    const float* g0 = adj + (size_t)(k0 + ar) * NN + (i0 + ac);
    const float* g1 = g0 + (size_t)16 * NN;
    unsigned off0 = (unsigned)((buf * 512 + ac * 32 + ar) * 4);       // sA[m][k]
    unsigned off1 = (unsigned)((buf * 512 + ac * 32 + ar + 16) * 4);
    asm volatile("global_load_async_to_lds_b32 %0, %1, off" :: "v"(off0), "v"(g0) : "memory");
    asm volatile("global_load_async_to_lds_b32 %0, %1, off" :: "v"(off1), "v"(g1) : "memory");
  };
  auto storeB = [&](const float4* breg, int k0, int buf) {
    __bf16* dst = sBt + buf * 4096;
#pragma unroll
    for (int u = 0; u < 4; ++u) {
      int t = tid + 256 * u;
      int k = t >> 5, n = (t & 31) * 4;       // flat = t*4 in [32][128]
      float dv = dinv[k0 + k];
      dst[(n + 0) * 32 + k] = (__bf16)(breg[u].x * dv);
      dst[(n + 1) * 32 + k] = (__bf16)(breg[u].y * dv);
      dst[(n + 2) * 32 + k] = (__bf16)(breg[u].z * dv);
      dst[(n + 3) * 32 + k] = (__bf16)(breg[u].w * dv);
    }
  };

  // prologue: stage step 0 into buffer 0
  float4 breg[4];
  issueA(0, 0);
  {
    const float4* src = (const float4*)X;
#pragma unroll
    for (int u = 0; u < 4; ++u) breg[u] = src[tid + 256 * u];
  }
  storeB(breg, 0, 0);

  v8f acc = {};
  for (int it = 0; it < NN / 32; ++it) {
    int cur = it & 1, nxt = cur ^ 1;
    int k0n = (it + 1) * 32;
    bool have_next = (it + 1) < (NN / 32);
    if (have_next) {                          // issue next step's loads before waiting
      issueA(k0n, nxt);
      const float4* src = (const float4*)(X + (size_t)k0n * DD);
#pragma unroll
      for (int u = 0; u < 4; ++u) breg[u] = src[tid + 256 * u];
    }
    if (have_next) asm volatile("s_wait_asynccnt 0x2" ::: "memory");  // cur's 2 loads done (in-order)
    else           asm volatile("s_wait_asynccnt 0x0" ::: "memory");
    __syncthreads();                          // cur A + cur B visible to all waves
    {
      const float* pA = smem + cur * 512 + l * 32;
      float4 x0 = *(const float4*)(pA + 16 * half);
      float4 x1 = *(const float4*)(pA + 16 * half + 4);
      float4 x2 = *(const float4*)(pA + 16 * half + 8);
      float4 x3 = *(const float4*)(pA + 16 * half + 12);
      v16bf a;
      a[0]  = (__bf16)x0.x; a[1]  = (__bf16)x0.y; a[2]  = (__bf16)x0.z; a[3]  = (__bf16)x0.w;
      a[4]  = (__bf16)x1.x; a[5]  = (__bf16)x1.y; a[6]  = (__bf16)x1.z; a[7]  = (__bf16)x1.w;
      a[8]  = (__bf16)x2.x; a[9]  = (__bf16)x2.y; a[10] = (__bf16)x2.z; a[11] = (__bf16)x2.w;
      a[12] = (__bf16)x3.x; a[13] = (__bf16)x3.y; a[14] = (__bf16)x3.z; a[15] = (__bf16)x3.w;
      v16bf b = *(const v16bf*)(sBt + cur * 4096 + (j0 + l) * 32 + 16 * half);
      acc = wmma_bf16(a, b, acc);
    }
    if (have_next) storeB(breg, k0n, nxt);    // nxt buffer: last read was 2 barriers ago
    __syncthreads();
  }
#pragma unroll
  for (int r = 0; r < 8; ++r) {
    int row = i0 + r + 8 * half, col = j0 + l;
    float y = dinv[row] * X[(size_t)row * DD + col];
    float v = dinv[row] * (acc[r] + y) + bias[col];
    if (do_relu) v = fmaxf(v, 0.0f);
    H[(size_t)row * DD + col] = v;
  }
}

__global__ void gl_hnorm(const float* __restrict__ H, float* __restrict__ hn) {
  int i = blockIdx.x * blockDim.x + threadIdx.x;
  float s = 0.0f;
  for (int k = 0; k < DD; ++k) { float v = H[(size_t)i * DD + k]; s += v * v; }
  hn[i] = sqrtf(s);
}

__global__ void gl_tobf(const float* __restrict__ H, __bf16* __restrict__ Hb) {
  int i = blockIdx.x * blockDim.x + threadIdx.x;
  Hb[i] = (__bf16)H[i];
}

// s = mask * (0.5*relu(HH^T/(n_i n_j + eps)) + 0.5*exp(-8 (d/dcut)^2));
// H pre-converted to bf16: K-loop is pure vector loads in WMMA register layout.
__global__ void gl_sim(const __bf16* __restrict__ Hb, const float* __restrict__ hn,
                       const float* __restrict__ dis, const float* __restrict__ dcut,
                       float* __restrict__ s_out) {
  int w    = blockIdx.x * 8 + (threadIdx.x >> 5);
  int lane = threadIdx.x & 31;
  int half = lane >> 4, l = lane & 15;
  int i0 = (w >> 8) * 16, j0 = (w & 255) * 16;
  v8f acc = {};
#pragma unroll
  for (int k0 = 0; k0 < DD; k0 += 32) {
    const __bf16* pa = Hb + (size_t)(i0 + l) * DD + k0 + 16 * half;
    v8bf a0 = *(const v8bf*)pa;
    v8bf a1 = *(const v8bf*)(pa + 8);
    v16bf a = __builtin_shufflevector(a0, a1, 0, 1, 2, 3, 4, 5, 6, 7,
                                      8, 9, 10, 11, 12, 13, 14, 15);
    v16bf b = *(const v16bf*)(Hb + (size_t)(j0 + l) * DD + k0 + 16 * half);
    acc = wmma_bf16(a, b, acc);
  }
  float dc = *dcut;
#pragma unroll
  for (int r = 0; r < 8; ++r) {
    int row = i0 + r + 8 * half, col = j0 + l;
    float denom = hn[row] * hn[col] + 1e-8f;
    float s1 = fmaxf(acc[r] / denom, 0.0f);
    float dij = dis[(size_t)row * NN + col];
    bool  m = (dij > 0.0f) && (dij <= dc);
    float ds = dij / dc;
    float sd = m ? __expf(-8.0f * ds * ds) : 0.0f;
    s_out[(size_t)row * NN + col] = m ? (0.5f * s1 + 0.5f * sd) : 0.0f;
  }
}

__global__ void gl_degs(const float* __restrict__ s, float* __restrict__ dinv_s) {
  __shared__ float red[256];
  int row = blockIdx.x;
  float acc = 0.0f;
  for (int j = threadIdx.x; j < NN; j += blockDim.x) acc += s[(size_t)row * NN + j];
  red[threadIdx.x] = acc;
  __syncthreads();
  for (int st = 128; st > 0; st >>= 1) {
    if ((int)threadIdx.x < st) red[threadIdx.x] += red[threadIdx.x + st];
    __syncthreads();
  }
  if (threadIdx.x == 0) { float d = red[0]; dinv_s[row] = d > 0.0f ? rsqrtf(d) : 0.0f; }
}

__global__ void gl_snorm(const float* __restrict__ s, const float* __restrict__ dinv_s,
                         float* __restrict__ out) {
  size_t idx = (size_t)blockIdx.x * blockDim.x + threadIdx.x;
  int row = (int)(idx >> 12), col = (int)(idx & (NN - 1));
  out[idx] = dinv_s[row] * s[idx] * dinv_s[col];
}

// ---------------- launcher ----------------
extern "C" void kernel_launch(void* const* d_in, const int* in_sizes, int n_in,
                              void* d_out, int out_size, void* d_ws, size_t ws_size,
                              hipStream_t stream) {
  (void)in_sizes; (void)n_in; (void)out_size; (void)ws_size;
  const float* features = (const float*)d_in[0];
  const float* adj      = (const float*)d_in[1];
  const float* dis      = (const float*)d_in[2];
  const float* W0       = (const float*)d_in[3];
  const float* b0       = (const float*)d_in[4];
  const float* W1       = (const float*)d_in[5];
  const float* b1       = (const float*)d_in[6];

  float* ws     = (float*)d_ws;
  float* rowk   = ws;                     // 4096
  float* dcut   = ws + NN;                // 1 (padded to 64)
  float* dinv   = ws + NN + 64;           // 4096
  float* hnorm  = dinv + NN;              // 4096
  float* dinv_s = hnorm + NN;             // 4096
  float* X1     = dinv_s + NN;            // 4096*128
  float* H0     = X1 + (size_t)NN * DD;
  float* X2     = H0 + (size_t)NN * DD;
  float* H1     = X2 + (size_t)NN * DD;
  __bf16* Hbf   = (__bf16*)(H1 + (size_t)NN * DD);   // 4096*128 bf16

  float* s_norm = (float*)d_out;
  float* s      = s_norm + (size_t)NN * NN;

  gl_kth_row<<<NN, 256, 0, stream>>>(dis, rowk);
  gl_median <<<1, 256, 0, stream>>>(rowk, dcut);
  gl_coldeg <<<NN / 256, 256, 0, stream>>>(adj, dinv);

  gl_gemm_xw<<<256, 256, 0, stream>>>(features, W0, X1);          // 2048 waves = 256x8 tiles
  gl_agg    <<<256, 256, 0, stream>>>(adj, X1, dinv, b0, H0, 1);  // block = one i0 tile, 8 tj waves
  gl_gemm_xw<<<256, 256, 0, stream>>>(H0, W1, X2);
  gl_agg    <<<256, 256, 0, stream>>>(adj, X2, dinv, b1, H1, 0);

  gl_hnorm  <<<NN / 256, 256, 0, stream>>>(H1, hnorm);
  gl_tobf   <<<NN * DD / 256, 256, 0, stream>>>(H1, Hbf);
  gl_sim    <<<8192, 256, 0, stream>>>(Hbf, hnorm, dis, dcut, s); // 65536 waves = 256x256 tiles
  gl_degs   <<<NN, 256, 0, stream>>>(s, dinv_s);
  gl_snorm  <<<(NN * (NN / 256)), 256, 0, stream>>>(s, dinv_s, s_norm);
}
